// AttentionBlock_14113262535368
// MI455X (gfx1250) — matrix operand order repaired and checked by reading, not compiled, not run on hardware
//
#include <hip/hip_runtime.h>

#define HW     16384
#define NSPLIT 32            // spatial chunks per batch for the GEMM kernels
#define SUBT   4             // 128-wide subtiles per chunk (NSPLIT*SUBT*128 == HW)

typedef __attribute__((ext_vector_type(16))) __bf16 v16bf;
typedef __attribute__((ext_vector_type(8)))  float  v8f;

union Frag {
    v16bf v;
    uint4 q[2];
};

__device__ __forceinline__ unsigned short f32_bf16(float f) {
    unsigned int u = __float_as_uint(f);
    u += 0x7FFFu + ((u >> 16) & 1u);   // round-to-nearest-even
    return (unsigned short)(u >> 16);
}

// ISA 10.2: flat->LDS aperture mapping takes the LDS address from the low 32 address bits, so the low
// 32 bits of a flat pointer to __shared__ memory are the DS byte address.
__device__ __forceinline__ unsigned lds_lo32(const void* p) {
    return (unsigned)(unsigned long long)p;
}

// CDNA5 async DMA: global -> LDS, 16B per lane, tracked by ASYNCcnt.
__device__ __forceinline__ void async_copy_b128(unsigned lds_byte, const void* gptr) {
    asm volatile("global_load_async_to_lds_b128 %0, %1, off"
                 :: "v"(lds_byte), "v"(gptr) : "memory");
}
__device__ __forceinline__ void wait_async0() {
    asm volatile("s_wait_asynccnt 0x0" ::: "memory");
}

// ---------------------------------------------------------------------------
// Kernel 1: GroupNorm partial stats. grid = 64 (b,g) * 32 chunks.
// Each (b,g) region is 16 ch * 16384 = 262144 contiguous floats.
// ---------------------------------------------------------------------------
__global__ __launch_bounds__(256) void k_stats(const float* __restrict__ x,
                                               float* __restrict__ statp) {
    __shared__ float ssum[256];
    __shared__ float ssq[256];
    const int t = threadIdx.x;
    const int bg = blockIdx.x >> 5;      // 0..63
    const int chunk = blockIdx.x & 31;   // 0..31
    const float4* base = (const float4*)(x + (size_t)bg * 262144 + (size_t)chunk * 8192);
    float sum = 0.f, sq = 0.f;
#pragma unroll
    for (int i = 0; i < 8; ++i) {
        float4 v = base[i * 256 + t];
        sum += v.x + v.y + v.z + v.w;
        sq  += v.x * v.x + v.y * v.y + v.z * v.z + v.w * v.w;
    }
    ssum[t] = sum; ssq[t] = sq;
    __syncthreads();
    for (int off = 128; off > 0; off >>= 1) {
        if (t < off) { ssum[t] += ssum[t + off]; ssq[t] += ssq[t + off]; }
        __syncthreads();
    }
    if (t == 0) {
        statp[blockIdx.x * 2 + 0] = ssum[0];
        statp[blockIdx.x * 2 + 1] = ssq[0];
    }
}

// ---------------------------------------------------------------------------
// Kernel 2: finalize stats, fold GN affine into conv_in weights (bf16) + bias.
// grid = 8 batches, 128 threads.
// ---------------------------------------------------------------------------
__global__ __launch_bounds__(128) void k_fold_in(const float* __restrict__ statp,
                                                 const float* __restrict__ gn_w,
                                                 const float* __restrict__ gn_b,
                                                 const float* __restrict__ w_in,
                                                 const float* __restrict__ b_in,
                                                 unsigned short* __restrict__ w1,
                                                 float* __restrict__ b1) {
    __shared__ float gm[8], gr[8];
    __shared__ float sa[128], sd[128];
    const int b = blockIdx.x;
    const int t = threadIdx.x;
    if (t < 8) {
        const float* p = statp + ((size_t)b * 8 + t) * 64;  // 32 chunks x {sum,sumsq}
        float sum = 0.f, sq = 0.f;
        for (int c2 = 0; c2 < 32; ++c2) { sum += p[c2 * 2]; sq += p[c2 * 2 + 1]; }
        const float inv = 1.0f / 262144.0f;
        float mean = sum * inv;
        float var  = sq * inv - mean * mean;
        gm[t] = mean;
        gr[t] = rsqrtf(var + 1e-5f);
    }
    __syncthreads();
    {
        const int g = t >> 4;
        float a = gn_w[t] * gr[g];
        sa[t] = a;
        sd[t] = gn_b[t] - gm[g] * a;
    }
    __syncthreads();
    const int o = t;
    float bias = b_in[o];
    unsigned short* wrow = w1 + ((size_t)b * 128 + o) * 128;
    for (int c = 0; c < 128; ++c) {
        float wv = w_in[o * 128 + c];
        wrow[c] = f32_bf16(wv * sa[c]);
        bias += wv * sd[c];
    }
    b1[b * 128 + o] = bias;
}

// ---------------------------------------------------------------------------
// Kernel 3: fused GN+conv_in GEMM (bf16 WMMA) + per-head Gram (scores) WMMA.
// grid = (NSPLIT, 8), 256 threads (8 waves). Wave w owns output rows
// [16w,16w+16) == head w. H1 (f32) is written into d_out.
// ---------------------------------------------------------------------------
__global__ __launch_bounds__(256) void k_gemm1(const float* __restrict__ x,
                                               const unsigned short* __restrict__ w1,
                                               const float* __restrict__ b1,
                                               float* __restrict__ h1,
                                               float* __restrict__ scorep) {
    __shared__ alignas(16) unsigned short sW[128 * 128];   // W1' [o][c]
    __shared__ alignas(16) unsigned short sXT[128 * 136];  // x tile transposed [hw][c]
    __shared__ alignas(16) unsigned short sH[128 * 136];   // H1 tile [o][hw] for Gram
    __shared__ float sB1[128];

    const int t = threadIdx.x;
    const int lane = t & 31;
    const int w = t >> 5;                 // wave id == head id
    const int ln = lane & 15;
    const int hi = (lane < 16) ? 0 : 1;   // lane half
    const int b = blockIdx.y;
    const int s = blockIdx.x;
    const int hw0 = s * (SUBT * 128);

    // stage folded weights (bf16) via CDNA5 async global->LDS DMA
    {
        const char* src = (const char*)(w1 + (size_t)b * 128 * 128);
        const unsigned dst = lds_lo32(sW);
#pragma unroll
        for (int r = 0; r < 8; ++r) {
            const unsigned off = (unsigned)(r * 256 + t) * 16u;
            async_copy_b128(dst + off, src + off);
        }
    }
    if (t < 128) sB1[t] = b1[b * 128 + t];

    v8f score = (v8f)0.0f;

    for (int sub = 0; sub < SUBT; ++sub) {
        const int hwt = hw0 + sub * 128;
        __syncthreads();  // sXT (and first-iter weight DMA) safe to (re)use
        // stage x tile, transposed, converting f32 -> bf16
        {
            const int cj = t >> 5;
            const int colb = (t & 31) * 4;
#pragma unroll
            for (int r = 0; r < 16; ++r) {
                const int c = r * 8 + cj;
                const float4 v = *(const float4*)(x + ((size_t)(b * 128 + c)) * HW + hwt + colb);
                sXT[(colb + 0) * 136 + c] = f32_bf16(v.x);
                sXT[(colb + 1) * 136 + c] = f32_bf16(v.y);
                sXT[(colb + 2) * 136 + c] = f32_bf16(v.z);
                sXT[(colb + 3) * 136 + c] = f32_bf16(v.w);
            }
        }
        // prefetch next subtile's x rows into cache while we compute
        if (sub + 1 < SUBT) {
            const int c = t & 127;
            const int half = t >> 7;
            __builtin_prefetch(x + ((size_t)(b * 128 + c)) * HW + hwt + 128 + half * 64, 0, 0);
        }
        wait_async0();     // weight DMA complete (no-op after first iteration)
        __syncthreads();

        v8f acc[8];
#pragma unroll
        for (int nt = 0; nt < 8; ++nt) acc[nt] = (v8f)0.0f;

#pragma unroll
        for (int ks = 0; ks < 4; ++ks) {
            Frag A;  // 16x32 bf16, rows = output channels of this wave, K = input ch
            const unsigned short* ap = sW + (16 * w + ln) * 128 + 32 * ks + (hi ? 8 : 0);
            A.q[0] = *(const uint4*)(ap);
            A.q[1] = *(const uint4*)(ap + 16);
#pragma unroll
            for (int nt = 0; nt < 8; ++nt) {
                Frag B;  // 32x16 bf16, K = input ch, N = hw
                const unsigned short* bp = sXT + (nt * 16 + ln) * 136 + 32 * ks + (hi ? 8 : 0);
                B.q[0] = *(const uint4*)(bp);
                B.q[1] = *(const uint4*)(bp + 16);
                acc[nt] = __builtin_amdgcn_wmma_f32_16x16x32_bf16(
                    false, A.v, false, B.v, (short)0, acc[nt], false, false);
            }
        }

        // epilogue: +bias, store H1 (f32, into d_out) + bf16 copy to LDS for Gram
#pragma unroll
        for (int nt = 0; nt < 8; ++nt) {
            const int col = nt * 16 + ln;
#pragma unroll
            for (int v = 0; v < 8; ++v) {
                const int m = 16 * w + (hi ? v + 8 : v);
                float val = acc[nt][v] + sB1[m];
                h1[((size_t)(b * 128 + m)) * HW + hwt + col] = val;
                sH[m * 136 + col] = f32_bf16(val);
            }
        }
        __syncthreads();

        // Gram for head w over this subtile's 128 spatial positions (K axis)
#pragma unroll
        for (int ks = 0; ks < 4; ++ks) {
            Frag A, B;
            const unsigned short* ap = sH + (16 * w + ln) * 136 + 32 * ks + (hi ? 8 : 0);
            A.q[0] = *(const uint4*)(ap);
            A.q[1] = *(const uint4*)(ap + 16);
            const unsigned short* bp = sH + (16 * w + ln) * 136 + 32 * ks + (hi ? 8 : 0);
            B.q[0] = *(const uint4*)(bp);
            B.q[1] = *(const uint4*)(bp + 16);
            score = __builtin_amdgcn_wmma_f32_16x16x32_bf16(
                false, A.v, false, B.v, (short)0, score, false, false);
        }
    }

    // deterministic score partials: [b][s][h*256 + i*16 + j]
    {
        float* sp = scorep + ((size_t)(b * NSPLIT + s)) * 2048 + w * 256;
#pragma unroll
        for (int v = 0; v < 8; ++v) {
            const int i = hi ? v + 8 : v;
            sp[i * 16 + ln] = score[v];
        }
    }
}

// ---------------------------------------------------------------------------
// Kernel 4: reduce score partials, softmax, fold weights into w_out -> M (bf16).
// grid = 8 batches, 256 threads.
// ---------------------------------------------------------------------------
__global__ __launch_bounds__(256) void k_softmax_fold(const float* __restrict__ scorep,
                                                      const float* __restrict__ w_out,
                                                      unsigned short* __restrict__ Mw) {
    __shared__ float sc[2048];  // [h][i][j]
    const int b = blockIdx.x;
    const int t = threadIdx.x;
#pragma unroll
    for (int r = 0; r < 8; ++r) {
        const int idx = r * 256 + t;
        float acc = 0.f;
        for (int s = 0; s < NSPLIT; ++s)
            acc += scorep[((size_t)(b * NSPLIT + s)) * 2048 + idx];
        sc[idx] = acc * 0.25f;  // HEAD_DIM^-0.5 = 1/4
    }
    __syncthreads();
    if (t < 128) {  // one (h,i) row per thread
        float* row = sc + t * 16;
        float m = row[0];
        for (int j = 1; j < 16; ++j) m = fmaxf(m, row[j]);
        float ssum = 0.f;
        for (int j = 0; j < 16; ++j) { float e = __expf(row[j] - m); row[j] = e; ssum += e; }
        float inv = 1.0f / ssum;
        for (int j = 0; j < 16; ++j) row[j] *= inv;
    }
    __syncthreads();
    // M[o][16h+j] = sum_i w_out[o][16h+i] * softmax[h][i][j]
#pragma unroll
    for (int k = 0; k < 64; ++k) {
        const int idx = k * 256 + t;       // 0..16383
        const int o = idx >> 7;
        const int cp = idx & 127;
        const int h = cp >> 4;
        const int j = cp & 15;
        float acc = 0.f;
        for (int i = 0; i < 16; ++i)
            acc += w_out[o * 128 + h * 16 + i] * sc[h * 256 + i * 16 + j];
        Mw[((size_t)b * 128 + o) * 128 + cp] = f32_bf16(acc);
    }
}

// ---------------------------------------------------------------------------
// Kernel 5: out = x + b_out + M @ H1   (H1 read from d_out, overwritten in place
// per disjoint spatial tile; read-before-write within each block).
// ---------------------------------------------------------------------------
__global__ __launch_bounds__(256) void k_gemm2(const float* __restrict__ x,
                                               const unsigned short* __restrict__ Mw,
                                               const float* __restrict__ b_out,
                                               float* __restrict__ out) {
    __shared__ alignas(16) unsigned short sM[128 * 128];   // M [o][c]
    __shared__ alignas(16) unsigned short sHT[128 * 136];  // H1 tile transposed [hw][c]
    __shared__ float sBo[128];

    const int t = threadIdx.x;
    const int lane = t & 31;
    const int w = t >> 5;
    const int ln = lane & 15;
    const int hi = (lane < 16) ? 0 : 1;
    const int b = blockIdx.y;
    const int s = blockIdx.x;
    const int hw0 = s * (SUBT * 128);

    // stage folded attention+conv_out matrix via async DMA
    {
        const char* src = (const char*)(Mw + (size_t)b * 128 * 128);
        const unsigned dst = lds_lo32(sM);
#pragma unroll
        for (int r = 0; r < 8; ++r) {
            const unsigned off = (unsigned)(r * 256 + t) * 16u;
            async_copy_b128(dst + off, src + off);
        }
    }
    if (t < 128) sBo[t] = b_out[t];

    for (int sub = 0; sub < SUBT; ++sub) {
        const int hwt = hw0 + sub * 128;
        __syncthreads();
        // stage H1 tile (f32 in d_out) transposed as bf16
        {
            const int cj = t >> 5;
            const int colb = (t & 31) * 4;
#pragma unroll
            for (int r = 0; r < 16; ++r) {
                const int c = r * 8 + cj;
                const float4 v = *(const float4*)(out + ((size_t)(b * 128 + c)) * HW + hwt + colb);
                sHT[(colb + 0) * 136 + c] = f32_bf16(v.x);
                sHT[(colb + 1) * 136 + c] = f32_bf16(v.y);
                sHT[(colb + 2) * 136 + c] = f32_bf16(v.z);
                sHT[(colb + 3) * 136 + c] = f32_bf16(v.w);
            }
        }
        // prefetch next subtile's H1 rows
        if (sub + 1 < SUBT) {
            const int c = t & 127;
            const int half = t >> 7;
            __builtin_prefetch(out + ((size_t)(b * 128 + c)) * HW + hwt + 128 + half * 64, 0, 0);
        }
        wait_async0();
        __syncthreads();

        v8f acc[8];
#pragma unroll
        for (int nt = 0; nt < 8; ++nt) acc[nt] = (v8f)0.0f;

#pragma unroll
        for (int ks = 0; ks < 4; ++ks) {
            Frag A;
            const unsigned short* ap = sM + (16 * w + ln) * 128 + 32 * ks + (hi ? 8 : 0);
            A.q[0] = *(const uint4*)(ap);
            A.q[1] = *(const uint4*)(ap + 16);
#pragma unroll
            for (int nt = 0; nt < 8; ++nt) {
                Frag B;
                const unsigned short* bp = sHT + (nt * 16 + ln) * 136 + 32 * ks + (hi ? 8 : 0);
                B.q[0] = *(const uint4*)(bp);
                B.q[1] = *(const uint4*)(bp + 16);
                acc[nt] = __builtin_amdgcn_wmma_f32_16x16x32_bf16(
                    false, A.v, false, B.v, (short)0, acc[nt], false, false);
            }
        }

        // epilogue: residual + bias, overwrite d_out
#pragma unroll
        for (int nt = 0; nt < 8; ++nt) {
            const int col = nt * 16 + ln;
#pragma unroll
            for (int v = 0; v < 8; ++v) {
                const int m = 16 * w + (hi ? v + 8 : v);
                const size_t idx = ((size_t)(b * 128 + m)) * HW + hwt + col;
                out[idx] = acc[nt][v] + sBo[m] + x[idx];
            }
        }
        __syncthreads();  // all writes done before next subtile re-stages sHT
    }
}

// ---------------------------------------------------------------------------
// Host launcher
// ---------------------------------------------------------------------------
extern "C" void kernel_launch(void* const* d_in, const int* in_sizes, int n_in,
                              void* d_out, int out_size, void* d_ws, size_t ws_size,
                              hipStream_t stream) {
    (void)in_sizes; (void)n_in; (void)out_size; (void)ws_size;
    const float* x     = (const float*)d_in[0];
    const float* gn_w  = (const float*)d_in[1];
    const float* gn_b  = (const float*)d_in[2];
    const float* w_in  = (const float*)d_in[3];
    const float* b_in  = (const float*)d_in[4];
    const float* w_out = (const float*)d_in[5];
    const float* b_out = (const float*)d_in[6];
    float* out = (float*)d_out;

    char* ws = (char*)d_ws;
    float* statp            = (float*)(ws);                       //  16 KB (64*32*2 f32)
    float* scorep           = (float*)(ws + 16384);               //   2 MB (8*32*2048 f32)
    float* b1               = (float*)(ws + 16384 + 2097152);     //   4 KB
    unsigned short* w1      = (unsigned short*)(ws + 16384 + 2097152 + 4096);            // 256 KB bf16
    unsigned short* Mw      = (unsigned short*)(ws + 16384 + 2097152 + 4096 + 262144);   // 256 KB bf16

    k_stats       <<<dim3(2048),       256, 0, stream>>>(x, statp);
    k_fold_in     <<<dim3(8),          128, 0, stream>>>(statp, gn_w, gn_b, w_in, b_in, w1, b1);
    k_gemm1       <<<dim3(NSPLIT, 8),  256, 0, stream>>>(x, w1, b1, out, scorep);
    k_softmax_fold<<<dim3(8),          256, 0, stream>>>(scorep, w_out, Mw);
    k_gemm2       <<<dim3(NSPLIT, 8),  256, 0, stream>>>(x, Mw, b_out, out);
}
